// GAT_15994458211103
// MI455X (gfx1250) — matrix operand order repaired and checked by reading, not compiled, not run on hardware
//
#include <hip/hip_runtime.h>
#include <hip/hip_bf16.h>
#include <cstdint>
#include <cstddef>

#define NNODES 100000
#define NEDGES 1600000
#define ETOT   (NEDGES + NNODES)

typedef __attribute__((ext_vector_type(16))) _Float16 v16h;
typedef __attribute__((ext_vector_type(8)))  _Float16 v8h;
typedef __attribute__((ext_vector_type(2)))  _Float16 h2v;
typedef __attribute__((ext_vector_type(8)))  float    v8f;

// ---------- helpers ----------
static __device__ __forceinline__ unsigned f2mono(float f) {
  unsigned u = __float_as_uint(f);
  return (u & 0x80000000u) ? ~u : (u | 0x80000000u);
}
static __device__ __forceinline__ float mono2f(unsigned u) {
  return (u & 0x80000000u) ? __uint_as_float(u & 0x7fffffffu)
                           : __uint_as_float(~u);
}
static __device__ __forceinline__ float lrelu(float v, float a) {
  return v > 0.f ? v : a * v;
}
// CDNA5 packed-f16 global atomic add (ISA 15.18.3 opcode 89), non-returning.
// One 4B atomic RMW updates two f16 channels -> halves atomic op count+bytes.
static __device__ __forceinline__ void atomic_pk_add_f16(_Float16* addr, unsigned data) {
  asm volatile("global_atomic_pk_add_f16 %0, %1, off"
               :: "v"(addr), "v"(data) : "memory");
}

// ---------- conversions ----------
__global__ void k_f32_to_f16(const float* __restrict__ in,
                             _Float16* __restrict__ out, int n) {
  int i = blockIdx.x * blockDim.x + threadIdx.x;
  if (i < n) out[i] = (_Float16)in[i];
}

// W [Cin x Cout] row-major f32  ->  Wt [Cout x Cin] f16 (transposed)
__global__ void k_transpose_w(const float* __restrict__ W,
                              _Float16* __restrict__ Wt, int Cin, int Cout) {
  int i = blockIdx.x * blockDim.x + threadIdx.x;
  if (i >= Cin * Cout) return;
  int k = i / Cout, c = i % Cout;
  Wt[c * Cin + k] = (_Float16)W[i];
}

// ---------- WMMA GEMM: Out[M x Nc] = X[M x K] * Wt[Nc x K]^T ----------
// One wave per 16x16 output tile; 4x v_wmma_f32_16x16x32_f16 over K=128.
// Dual epilogue: f32 (for attention logits) + f16 (for the edge gather).
__global__ void k_gemm_wmma(const _Float16* __restrict__ X,
                            const _Float16* __restrict__ Wt,
                            float* __restrict__ Out,
                            _Float16* __restrict__ Out16,
                            int K, int Nc) {
  int tileM = blockIdx.x;
  int tileN = blockIdx.y;
  int lane  = threadIdx.x & 31;
  int half  = lane >> 4;   // 0: lanes 0-15, 1: lanes 16-31
  int l16   = lane & 15;

  const _Float16* xrow = X  + (size_t)(tileM * 16 + l16) * K;  // A row (M = l16)
  const _Float16* wrow = Wt + (size_t)(tileN * 16 + l16) * K;  // B col (N = l16)

  v8f acc = {};
  for (int ko = 0; ko < K; ko += 32) {
    // A layout (ISA 7.12.2): lanes 0-15 carry K = ko+0..7 and ko+16..23,
    // lanes 16-31 carry K = ko+8..15 and ko+24..31.
    int ka = ko + half * 8;
    v8h a0 = *(const v8h*)(xrow + ka);        // 16B
    v8h a1 = *(const v8h*)(xrow + ka + 16);   // 16B
    v16h a = __builtin_shufflevector(a0, a1,
               0,1,2,3,4,5,6,7,8,9,10,11,12,13,14,15);
    // B layout: same K order as A (lanes 0-15 K = ko+0..7,16..23; lanes 16-31 K = ko+8..15,24..31).
    v8h b0 = *(const v8h*)(wrow + ka);        // 16B
    v8h b1 = *(const v8h*)(wrow + ka + 16);   // 16B
    v16h b = __builtin_shufflevector(b0, b1,
               0,1,2,3,4,5,6,7,8,9,10,11,12,13,14,15);
    acc = __builtin_amdgcn_wmma_f32_16x16x32_f16(
        /*neg_a=*/false, a, /*neg_b=*/false, b,
        /*c_mod=*/(short)0, acc, /*reuse_a=*/false, /*reuse_b=*/false);
  }
  // C/D layout: VGPR r -> row (r + half*8), col l16
  int col = tileN * 16 + l16;
  size_t base = (size_t)(tileM * 16 + half * 8) * Nc + col;
  float*    o32 = Out   + base;
  _Float16* o16 = Out16 + base;
  #pragma unroll
  for (int r = 0; r < 8; ++r) {
    o32[(size_t)r * Nc] = acc[r];
    o16[(size_t)r * Nc] = (_Float16)acc[r];
  }
}

// ---------- attention logits: als/ald [N x H] ----------
__global__ void k_logits(const float* __restrict__ proj,
                         const float* __restrict__ a_src,
                         const float* __restrict__ a_dst,
                         float* __restrict__ als, float* __restrict__ ald,
                         int H, int HD) {
  int i = blockIdx.x * blockDim.x + threadIdx.x;
  if (i >= NNODES * H) return;
  int n = i / H, h = i % H;
  const float* p  = proj + (size_t)n * (H * HD) + h * HD;
  const float* as = a_src + h * HD;
  const float* ad = a_dst + h * HD;
  float ss = 0.f, sd = 0.f;
  for (int c = 0; c < HD; ++c) { float v = p[c]; ss += v * as[c]; sd += v * ad[c]; }
  als[i] = ss; ald[i] = sd;
}

// ---------- per-call state init (acc16=0, m=-inf(mono), s=0) ----------
__global__ void k_init(unsigned* __restrict__ acc16w, int nacc,   // f16 accum as u32 words
                       unsigned* __restrict__ m, float* __restrict__ s, int nms) {
  int i = blockIdx.x * blockDim.x + threadIdx.x;
  if (i < nacc) acc16w[i] = 0u;
  if (i < nms) { m[i] = 0x007fffffu; s[i] = 0.f; }  // f2mono(-inf)
}

static __device__ __forceinline__ void edge_sd(const int* ei, int e, int& s_, int& d_) {
  if (e < NEDGES) { s_ = ei[e]; d_ = ei[NEDGES + e]; }
  else            { s_ = d_ = e - NEDGES; }          // self-loops appended
}

// ---------- edge pass 1: segment max ----------
__global__ void k_edge_max(const int* __restrict__ ei,
                           const float* __restrict__ als,
                           const float* __restrict__ ald,
                           unsigned* __restrict__ m, int H) {
  int i = blockIdx.x * blockDim.x + threadIdx.x;
  if (i >= ETOT * H) return;
  int e = i / H, h = i % H;
  int s_, d_; edge_sd(ei, e, s_, d_);
  float v = lrelu(als[s_ * H + h] + ald[d_ * H + h], 0.2f);
  atomicMax(&m[d_ * H + h], f2mono(v));
}

// ---------- edge pass 2: segment sum of exp ----------
__global__ void k_edge_sum(const int* __restrict__ ei,
                           const float* __restrict__ als,
                           const float* __restrict__ ald,
                           const unsigned* __restrict__ m,
                           float* __restrict__ s, int H) {
  int i = blockIdx.x * blockDim.x + threadIdx.x;
  if (i >= ETOT * H) return;
  int e = i / H, h = i % H;
  int s_, d_; edge_sd(ei, e, s_, d_);
  float v = lrelu(als[s_ * H + h] + ald[d_ * H + h], 0.2f);
  float p = __expf(v - mono2f(m[d_ * H + h]));
  atomicAdd(&s[d_ * H + h], p);
}

// ---------- edge pass 3: weighted feature scatter (wave per edge) ----------
// f16 gather + packed-f16 atomics: half the bytes, half the atomic ops of f32.
__global__ void k_edge_aggr(const int* __restrict__ ei,
                            const float* __restrict__ als,
                            const float* __restrict__ ald,
                            const unsigned* __restrict__ m,
                            const float* __restrict__ s,
                            const _Float16* __restrict__ proj16,
                            _Float16* __restrict__ acc16, int H, int HD) {
  int gid  = blockIdx.x * blockDim.x + threadIdx.x;
  int e    = gid >> 5;
  int lane = gid & 31;
  if (e >= ETOT) return;
  int s_, d_; edge_sd(ei, e, s_, d_);
  int Cout = H * HD;
  const _Float16* ps = proj16 + (size_t)s_ * Cout;
  _Float16*       pd = acc16  + (size_t)d_ * Cout;
  __builtin_prefetch(ps + lane * 2, 0, 0);   // global_prefetch_b8
  // each lane owns channel pair (2*lane, 2*lane+1); pairs never straddle heads
  for (int c = lane * 2; c < Cout; c += 64) {
    int h = c / HD;
    float v = lrelu(als[s_ * H + h] + ald[d_ * H + h], 0.2f);
    float p = __expf(v - mono2f(m[d_ * H + h]));
    float alpha = p / (s[d_ * H + h] + 1e-16f);
    h2v pv = *(const h2v*)(ps + c);
    h2v prod;
    prod.x = (_Float16)(alpha * (float)pv.x);
    prod.y = (_Float16)(alpha * (float)pv.y);
    unsigned data;
    __builtin_memcpy(&data, &prod, 4);
    atomic_pk_add_f16(pd + c, data);         // global_atomic_pk_add_f16
  }
}

// ---------- epilogues (wave per node) ----------
// layers 0/1: v = leaky(acc16+b, 0.01); optional L1-norm; emit f16 for next GEMM
__global__ void k_epi01(const _Float16* __restrict__ acc16,
                        const float* __restrict__ bias,
                        _Float16* __restrict__ out16, int Cout, int do_l1) {
  int gid = blockIdx.x * blockDim.x + threadIdx.x;
  int n = gid >> 5, lane = gid & 31;
  if (n >= NNODES) return;
  const _Float16* a = acc16 + (size_t)n * Cout;
  _Float16*       o = out16 + (size_t)n * Cout;
  float vals[4];
  int per = Cout >> 5;               // <= 4
  float asum = 0.f;
  for (int j = 0; j < per; ++j) {
    int c = lane + j * 32;
    float v = lrelu((float)a[c] + bias[c], 0.01f);
    vals[j] = v;
    asum += fabsf(v);
  }
  if (do_l1) {
    for (int off = 16; off > 0; off >>= 1) asum += __shfl_xor(asum, off, 32);
    float inv = 1.f / fmaxf(asum, 1e-12f);
    for (int j = 0; j < per; ++j) vals[j] *= inv;
  }
  for (int j = 0; j < per; ++j) o[lane + j * 32] = (_Float16)vals[j];
}

// layer 2: out = relu(l1norm(acc16+b)), f32 to d_out
__global__ void k_epi2(const _Float16* __restrict__ acc16,
                       const float* __restrict__ bias,
                       float* __restrict__ out, int Cout) {
  int gid = blockIdx.x * blockDim.x + threadIdx.x;
  int n = gid >> 5, lane = gid & 31;
  if (n >= NNODES) return;
  const _Float16* a = acc16 + (size_t)n * Cout;
  float*          o = out   + (size_t)n * Cout;
  float vals[2];
  int per = Cout >> 5;               // 2
  float asum = 0.f;
  for (int j = 0; j < per; ++j) {
    float v = (float)a[lane + j * 32] + bias[lane + j * 32];
    vals[j] = v;
    asum += fabsf(v);
  }
  for (int off = 16; off > 0; off >>= 1) asum += __shfl_xor(asum, off, 32);
  float inv = 1.f / fmaxf(asum, 1e-12f);
  for (int j = 0; j < per; ++j)
    o[lane + j * 32] = fmaxf(vals[j] * inv, 0.f);
}

// ---------- host orchestration ----------
static inline int cdiv(long long a, long long b) { return (int)((a + b - 1) / b); }

extern "C" void kernel_launch(void* const* d_in, const int* in_sizes, int n_in,
                              void* d_out, int out_size, void* d_ws, size_t ws_size,
                              hipStream_t stream) {
  (void)in_sizes; (void)n_in; (void)out_size; (void)ws_size;
  const float* x   = (const float*)d_in[0];
  const int*   ei  = (const int*)  d_in[1];
  const float* W[3]  = { (const float*)d_in[2],  (const float*)d_in[6],  (const float*)d_in[10] };
  const float* B[3]  = { (const float*)d_in[3],  (const float*)d_in[7],  (const float*)d_in[11] };
  const float* AS[3] = { (const float*)d_in[4],  (const float*)d_in[8],  (const float*)d_in[12] };
  const float* AD[3] = { (const float*)d_in[5],  (const float*)d_in[9],  (const float*)d_in[13] };
  const int Cin[3]  = { 128, 128, 128 };
  const int Cout[3] = { 128, 128, 64 };
  const int Hh[3]   = { 4, 4, 1 };

  // workspace layout (256B aligned)
  char* ws = (char*)d_ws; size_t off = 0;
  auto take = [&](size_t bytes) -> void* {
    void* p = ws + off; off += (bytes + 255) & ~(size_t)255; return p;
  };
  float*    proj   = (float*)   take((size_t)NNODES * 128 * 4);  // f32 (logits)
  _Float16* proj16 = (_Float16*)take((size_t)NNODES * 128 * 2);  // f16 (gather)
  _Float16* acc16  = (_Float16*)take((size_t)NNODES * 128 * 2);  // f16 accumulator
  _Float16* x16    = (_Float16*)take((size_t)NNODES * 128 * 2);
  _Float16* wt16   = (_Float16*)take((size_t)128 * 128 * 2);
  float*    als    = (float*)   take((size_t)NNODES * 4 * 4);
  float*    ald    = (float*)   take((size_t)NNODES * 4 * 4);
  unsigned* mbuf   = (unsigned*)take((size_t)NNODES * 4 * 4);
  float*    sbuf   = (float*)   take((size_t)NNODES * 4 * 4);

  const int TPB = 256;
  // layer 0 input: convert x to f16 once
  k_f32_to_f16<<<cdiv((long long)NNODES * 128, TPB), TPB, 0, stream>>>(x, x16, NNODES * 128);

  for (int L = 0; L < 3; ++L) {
    int ci = Cin[L], co = Cout[L], H = Hh[L], HD = co / H;
    // transpose+convert weights: Wt [co x ci] f16
    k_transpose_w<<<cdiv(ci * co, TPB), TPB, 0, stream>>>(W[L], wt16, ci, co);
    // projection: proj(.f32/.f16) = x16 @ W
    k_gemm_wmma<<<dim3(NNODES / 16, co / 16), 32, 0, stream>>>(x16, wt16, proj, proj16, ci, co);
    // attention logits (f32 path)
    k_logits<<<cdiv((long long)NNODES * H, TPB), TPB, 0, stream>>>(proj, AS[L], AD[L], als, ald, H, HD);
    // reset acc16 / m / s (graph-replayed: must self-initialize every call)
    k_init<<<cdiv((long long)NNODES * co / 2, TPB), TPB, 0, stream>>>(
        (unsigned*)acc16, NNODES * co / 2, mbuf, sbuf, NNODES * H);
    // edge softmax (3 passes)
    k_edge_max <<<cdiv((long long)ETOT * H, TPB), TPB, 0, stream>>>(ei, als, ald, mbuf, H);
    k_edge_sum <<<cdiv((long long)ETOT * H, TPB), TPB, 0, stream>>>(ei, als, ald, mbuf, sbuf, H);
    k_edge_aggr<<<cdiv((long long)ETOT * 32, TPB), TPB, 0, stream>>>(ei, als, ald, mbuf, sbuf,
                                                                     proj16, acc16, H, HD);
    // epilogue
    if (L < 2) {
      k_epi01<<<cdiv((long long)NNODES * 32, TPB), TPB, 0, stream>>>(acc16, B[L], x16, co,
                                                                     /*do_l1=*/L == 0 ? 1 : 0);
    } else {
      k_epi2<<<cdiv((long long)NNODES * 32, TPB), TPB, 0, stream>>>(acc16, B[L], (float*)d_out, co);
    }
  }
}
